// GAT_58926951301825
// MI455X (gfx1250) — matrix operand order repaired and checked by reading, not compiled, not run on hardware
//
#include <hip/hip_runtime.h>
#include <hip/hip_bf16.h>
#include <cmath>

#define N_NODES   50000
#define N_EDGES   800000
#define N_ETOT    (N_EDGES + N_NODES)   // with self loops
#define F_IN      128
#define H1_DIM    256
#define H2_DIM    32
#define C_OUT     40
#define GAT_SLOPE 0.2f
#define ACT_SLOPE 0.01f

typedef __bf16 bf16_t;
typedef __attribute__((ext_vector_type(16))) __bf16 v16bf;
typedef __attribute__((ext_vector_type(8)))  __bf16 v8bf;
typedef __attribute__((ext_vector_type(8)))  float  v8f;

// ---------------------------------------------------------------- helpers ----
__device__ __forceinline__ void edge_sd(const int* __restrict__ ei, int e, int& s, int& d) {
    if (e < N_EDGES) { s = ei[e]; d = ei[N_EDGES + e]; }
    else             { s = e - N_EDGES; d = s; }          // self loop
}

__device__ __forceinline__ float leaky(float x, float slope) {
    return x > 0.f ? x : x * slope;
}

// float atomic max via signed-max / unsigned-min bit trick (init to -inf)
__device__ __forceinline__ void atomicMaxF(float* addr, float value) {
    if (value >= 0.f)
        atomicMax((int*)addr, __float_as_int(value));
    else
        atomicMin((unsigned int*)addr, __float_as_uint(value));
}

// ---------------------------------------------------------------- kernels ----
__global__ void fill_f32_kernel(float* __restrict__ p, float v, int n) {
    int i = blockIdx.x * blockDim.x + threadIdx.x;
    if (i < n) p[i] = v;
}

__global__ void f32_to_bf16_kernel(const float* __restrict__ in, bf16_t* __restrict__ out, int n) {
    int i = blockIdx.x * blockDim.x + threadIdx.x;
    if (i < n) out[i] = (bf16_t)in[i];
}

// W (K x N, f32 row-major) -> Wt (Npad x K, bf16 row-major), zero-padded columns.
// Transposed + padded so GEMM B-operand loads are contiguous and unconditional.
__global__ void conv_w_kernel(const float* __restrict__ W, bf16_t* __restrict__ Wt,
                              int K, int N, int Npad) {
    int i = blockIdx.x * blockDim.x + threadIdx.x;   // over Npad*K
    if (i >= Npad * K) return;
    int n = i / K;
    int k = i - n * K;
    Wt[i] = (n < N) ? (bf16_t)W[(size_t)k * N + n] : (bf16_t)0.0f;
}

// One wave computes a 16 x (16*NT) strip of C = A(MxK) * B(KxN), bf16 in, f32 accum.
// One A fragment is reused across NT WMMAs per K-step (NT x operand reuse).
// B supplied pre-transposed+padded as Bt (Npad x K row-major).
// VGPR layouts per CDNA5 ISA 7.12.2 (wave32):
//   A 16x32 bf16 : lane = half*16 + m ; elems 0..7 -> K = k0 + half*8 + i,
//                                       elems 8..15 -> K = k0 + 16 + half*8 + (i-8)
//   B 32x16 bf16 : lane = half*16 + n ; elem i -> same K as A elem i
//   C 16x16 f32  : lane = half*16 + n ; vgpr r -> M = half*8 + r
template<int NT>
__global__ void wmma_gemm_kernel(const bf16_t* __restrict__ A,
                                 const bf16_t* __restrict__ Bt,
                                 float* __restrict__ C,
                                 int M, int N, int K) {
    const int tileM  = blockIdx.x;          // M/16 exact (50000 = 3125*16)
    const int tileN0 = blockIdx.y * NT;     // first of NT consecutive N-tiles
    const int lane   = threadIdx.x & 31;
    const int half   = lane >> 4;           // 0|1
    const int l15    = lane & 15;
    const int m      = tileM * 16 + l15;    // A row held by this lane

    const bf16_t* __restrict__ arow = A + (size_t)m * K;
    const bf16_t* brow[NT];
    #pragma unroll
    for (int t = 0; t < NT; ++t)
        brow[t] = Bt + (size_t)((tileN0 + t) * 16 + l15) * K;   // always < Npad rows

    v8f acc[NT];
    #pragma unroll
    for (int t = 0; t < NT; ++t) acc[t] = (v8f){};

    for (int k0 = 0; k0 < K; k0 += 32) {
        // A: two contiguous 16B chunks (8 bf16 each), 16B aligned -- loaded once per K-step
        v8bf a0 = *(const v8bf*)(arow + k0 + half * 8);
        v8bf a1 = *(const v8bf*)(arow + k0 + half * 8 + 16);
        __builtin_prefetch(arow + k0 + 64, 0, 1);   // global_prefetch_b8 on the streamed operand

        v16bf a;
        #pragma unroll
        for (int i = 0; i < 8; ++i) { a[i] = a0[i]; a[i + 8] = a1[i]; }

        #pragma unroll
        for (int t = 0; t < NT; ++t) {
            // B: two contiguous 16B chunks (8 bf16 each), same K order as A, 16B aligned
            v8bf b0 = *(const v8bf*)(brow[t] + k0 + half * 8);
            v8bf b1 = *(const v8bf*)(brow[t] + k0 + half * 8 + 16);
            v16bf b;
            #pragma unroll
            for (int i = 0; i < 8; ++i) { b[i] = b0[i]; b[i + 8] = b1[i]; }
            acc[t] = __builtin_amdgcn_wmma_f32_16x16x32_bf16(
                         /*neg_a=*/false, a, /*neg_b=*/false, b,
                         /*c_mod=*/(short)0, acc[t], /*reuse_a=*/false, /*reuse_b=*/false);
        }
    }

    #pragma unroll
    for (int t = 0; t < NT; ++t) {
        int n = (tileN0 + t) * 16 + l15;
        if (n < N) {
            #pragma unroll
            for (int r = 0; r < 8; ++r) {
                int mo = tileM * 16 + half * 8 + r;
                C[(size_t)mo * N + n] = acc[t][r];
            }
        }
    }
}

// One wave per node: alpha_s[i] = dot(H[i,:], a_s) ; alpha_d[i] = dot(H[i,:], a_d)
__global__ void alphas_kernel(const float* __restrict__ H,
                              const float* __restrict__ a_s,
                              const float* __restrict__ a_d,
                              float* __restrict__ out_s,
                              float* __restrict__ out_d,
                              int M, int N) {
    int node = blockIdx.x * (blockDim.x >> 5) + (threadIdx.x >> 5);
    int lane = threadIdx.x & 31;
    if (node >= M) return;
    float ss = 0.f, sd = 0.f;
    for (int c = lane; c < N; c += 32) {
        float h = H[(size_t)node * N + c];
        ss += h * a_s[c];
        sd += h * a_d[c];
    }
    #pragma unroll
    for (int off = 16; off; off >>= 1) {
        ss += __shfl_xor(ss, off, 32);
        sd += __shfl_xor(sd, off, 32);
    }
    if (lane == 0) { out_s[node] = ss; out_d[node] = sd; }
}

// pass 1: segment max of leaky(as[src]+ad[dst]) over dst
__global__ void edge_max_kernel(const int* __restrict__ ei,
                                const float* __restrict__ as_, const float* __restrict__ ad_,
                                float* __restrict__ mx) {
    int e = blockIdx.x * blockDim.x + threadIdx.x;
    if (e >= N_ETOT) return;
    int s, d; edge_sd(ei, e, s, d);
    float l = leaky(as_[s] + ad_[d], GAT_SLOPE);
    atomicMaxF(&mx[d], l);
}

// pass 2: denom[d] += exp(logit - m[d])
__global__ void edge_sum_kernel(const int* __restrict__ ei,
                                const float* __restrict__ as_, const float* __restrict__ ad_,
                                const float* __restrict__ mx, float* __restrict__ denom) {
    int e = blockIdx.x * blockDim.x + threadIdx.x;
    if (e >= N_ETOT) return;
    int s, d; edge_sd(ei, e, s, d);
    float l = leaky(as_[s] + ad_[d], GAT_SLOPE);
    atomicAdd(&denom[d], __expf(l - mx[d]));
}

// pass 3: out[d,:] += (exp(logit-m[d])/denom[d]) * H[s,:]   (one wave per edge)
__global__ void edge_scatter_kernel(const int* __restrict__ ei,
                                    const float* __restrict__ H,
                                    const float* __restrict__ as_, const float* __restrict__ ad_,
                                    const float* __restrict__ mx, const float* __restrict__ denom,
                                    float* __restrict__ out, int N) {
    int e    = blockIdx.x * (blockDim.x >> 5) + (threadIdx.x >> 5);
    int lane = threadIdx.x & 31;
    if (e >= N_ETOT) return;
    int s, d; edge_sd(ei, e, s, d);
    float l = leaky(as_[s] + ad_[d], GAT_SLOPE);
    float w = __expf(l - mx[d]) / denom[d];
    for (int c = lane; c < N; c += 32)
        atomicAdd(&out[(size_t)d * N + c], w * H[(size_t)s * N + c]);
}

// out += bias ; optional leaky-relu ; optional bf16 copy for next layer's GEMM input
__global__ void finalize_kernel(float* __restrict__ out, const float* __restrict__ bias,
                                bf16_t* __restrict__ nextb, int M, int N, int act) {
    int i = blockIdx.x * blockDim.x + threadIdx.x;
    if (i >= M * N) return;
    float v = out[i] + bias[i % N];
    if (act) v = leaky(v, ACT_SLOPE);
    out[i] = v;
    if (nextb) nextb[i] = (bf16_t)v;
}

// ------------------------------------------------------------------ host ----
extern "C" void kernel_launch(void* const* d_in, const int* in_sizes, int n_in,
                              void* d_out, int out_size, void* d_ws, size_t ws_size,
                              hipStream_t stream) {
    (void)in_sizes; (void)n_in; (void)out_size; (void)ws_size;

    const float* x   = (const float*)d_in[0];
    const int*   ei  = (const int*)  d_in[1];
    const float* W1  = (const float*)d_in[2];
    const float* a1s = (const float*)d_in[3];
    const float* a1d = (const float*)d_in[4];
    const float* b1  = (const float*)d_in[5];
    const float* W2  = (const float*)d_in[6];
    const float* a2s = (const float*)d_in[7];
    const float* a2d = (const float*)d_in[8];
    const float* b2  = (const float*)d_in[9];
    const float* W3  = (const float*)d_in[10];
    const float* a3s = (const float*)d_in[11];
    const float* a3d = (const float*)d_in[12];
    const float* b3  = (const float*)d_in[13];
    float* outp = (float*)d_out;

    // workspace carve-out
    char* ws = (char*)d_ws;
    size_t off = 0;
    auto carve = [&](size_t bytes) -> void* {
        off = (off + 255) & ~(size_t)255;
        void* p = ws + off;
        off += bytes;
        return p;
    };
    bf16_t* xb   = (bf16_t*)carve((size_t)N_NODES * H1_DIM * sizeof(bf16_t)); // layer input, bf16
    bf16_t* Wt   = (bf16_t*)carve((size_t)F_IN * H1_DIM * sizeof(bf16_t));    // weights, bf16, transposed+padded
    float*  h    = (float*) carve((size_t)N_NODES * H1_DIM * sizeof(float));  // GEMM out
    float*  agg  = (float*) carve((size_t)N_NODES * H1_DIM * sizeof(float));  // aggregation out (layers 1,2)
    float*  as_  = (float*) carve((size_t)N_NODES * sizeof(float));
    float*  ad_  = (float*) carve((size_t)N_NODES * sizeof(float));
    float*  mx   = (float*) carve((size_t)N_NODES * sizeof(float));
    float*  dnm  = (float*) carve((size_t)N_NODES * sizeof(float));

    const int B = 256;
    auto cdiv = [](int a, int b) { return (a + b - 1) / b; };
    const float NEG_INF = -__builtin_huge_valf();

    auto run_layer = [&](int K, int N, const float* W, const float* avs, const float* avd,
                         const float* bias, float* out_l, bf16_t* next_bf16, int act) {
        const int Npad = cdiv(N, 16) * 16;
        // W -> bf16, transposed (Npad x K), zero-padded
        conv_w_kernel<<<cdiv(Npad * K, B), B, 0, stream>>>(W, Wt, K, N, Npad);
        // h = xb @ W  (WMMA bf16 -> f32); NT N-tiles per wave for A-fragment reuse
        const int ntiles = Npad / 16;
        if (ntiles % 4 == 0) {
            dim3 g(N_NODES / 16, ntiles / 4);
            wmma_gemm_kernel<4><<<g, 32, 0, stream>>>(xb, Wt, h, N_NODES, N, K);
        } else if (ntiles % 3 == 0) {
            dim3 g(N_NODES / 16, ntiles / 3);
            wmma_gemm_kernel<3><<<g, 32, 0, stream>>>(xb, Wt, h, N_NODES, N, K);
        } else {
            dim3 g(N_NODES / 16, ntiles / 2);
            wmma_gemm_kernel<2><<<g, 32, 0, stream>>>(xb, Wt, h, N_NODES, N, K);
        }
        // per-node attention projections
        alphas_kernel<<<cdiv(N_NODES, B / 32), B, 0, stream>>>(h, avs, avd, as_, ad_, N_NODES, N);
        // segment softmax + aggregate
        fill_f32_kernel<<<cdiv(N_NODES, B), B, 0, stream>>>(mx, NEG_INF, N_NODES);
        fill_f32_kernel<<<cdiv(N_NODES, B), B, 0, stream>>>(dnm, 0.f, N_NODES);
        fill_f32_kernel<<<cdiv(N_NODES * N, B), B, 0, stream>>>(out_l, 0.f, N_NODES * N);
        edge_max_kernel<<<cdiv(N_ETOT, B), B, 0, stream>>>(ei, as_, ad_, mx);
        edge_sum_kernel<<<cdiv(N_ETOT, B), B, 0, stream>>>(ei, as_, ad_, mx, dnm);
        edge_scatter_kernel<<<cdiv(N_ETOT, B / 32), B, 0, stream>>>(ei, h, as_, ad_, mx, dnm, out_l, N);
        // bias + activation (+ bf16 copy for next layer)
        finalize_kernel<<<cdiv(N_NODES * N, B), B, 0, stream>>>(out_l, bias, next_bf16, N_NODES, N, act);
    };

    // layer 1: x(f32) -> bf16 input
    f32_to_bf16_kernel<<<cdiv(N_NODES * F_IN, B), B, 0, stream>>>(x, xb, N_NODES * F_IN);
    run_layer(F_IN,   H1_DIM, W1, a1s, a1d, b1, agg,  xb,      1);
    // layer 2: input = bf16(leaky(agg)) already in xb
    run_layer(H1_DIM, H2_DIM, W2, a2s, a2d, b2, agg,  xb,      1);
    // layer 3: final output, no activation
    run_layer(H2_DIM, C_OUT,  W3, a3s, a3d, b3, outp, nullptr, 0);
}
